// QGraphNetwork_5660766896163
// MI455X (gfx1250) — matrix operand order repaired and checked by reading, not compiled, not run on hardware
//
#include <hip/hip_runtime.h>
#include <hip/hip_bf16.h>

// ---------------- problem constants (from reference) ----------------
#define NBv 483328
#define NLv 761856
#define NDv 405504
#define NGv 24576
#define Nv  (NBv + NLv + NDv + NGv)   // 1,675,264
#define Ev  3276800
#define GEN_BASE (Nv - NGv)
#define Gv  4096

// flat param-buffer offsets (dict order: emb_bus, emb_line, emb_load, emb_gen, conv, val; W then b)
#define P_BUS_W   0
#define P_BUS_B   96
#define P_LINE_W  112
#define P_LINE_B  240
#define P_LOAD_W  256
#define P_LOAD_B  320
#define P_GEN_W   336
#define P_GEN_B   512
#define P_CONV_W  528
#define P_CONV_B  1040
#define P_VAL_W   1056
#define P_VAL_B   1248

typedef __attribute__((ext_vector_type(16))) __bf16 v16bf;
typedef __attribute__((ext_vector_type(8)))  float  v8f;

// sum arrays have NGv+1 rows: row NGv is a write-only dump row for padded tile slots
#define SUMROWS (NGv + 1)

// ---------------- K0: zero scratch (sums, counts, edge counter) ----------------
__global__ void ws_zero(unsigned* __restrict__ p, int n) {
  int i = blockIdx.x * blockDim.x + threadIdx.x;
  if (i < n) p[i] = 0u;
}

// ---------------- K1: embed gen nodes (both branches); also the skip ----------------
__global__ void embed_gen(const float* __restrict__ xgen, const float* __restrict__ action,
                          const float* __restrict__ p1, const float* __restrict__ p2,
                          float* __restrict__ xg1, float* __restrict__ xg2) {
  int tid = blockIdx.x * blockDim.x + threadIdx.x;
  if (tid >= NGv * 16) return;
  int n = tid >> 4, o = tid & 15;
  float f[11];
#pragma unroll
  for (int k = 0; k < 10; ++k) f[k] = xgen[n * 10 + k];
  f[10] = action[n];

  float a1 = p1[P_GEN_B + o];
  float a2 = p2[P_GEN_B + o];
#pragma unroll
  for (int k = 0; k < 11; ++k) {
    a1 = fmaf(p1[P_GEN_W + o * 11 + k], f[k], a1);
    a2 = fmaf(p2[P_GEN_W + o * 11 + k], f[k], a2);
  }
  xg1[tid] = a1;
  xg2[tid] = a2;
}

// ---------------- K2: compact edges whose dst is a gen node; count per dst ----------------
__global__ void edge_filter(const int* __restrict__ ei, int* __restrict__ pairs,
                            unsigned* __restrict__ ecnt, unsigned* __restrict__ cnt) {
  int e = blockIdx.x * blockDim.x + threadIdx.x;
  if (e >= Ev) return;
  __builtin_prefetch(&ei[Ev + e + 8192], 0, 1);   // global_prefetch_b8 stream-ahead
  int dst = ei[Ev + e];                           // edge_index[1][e]
  if (dst >= GEN_BASE) {
    int src = ei[e];                              // edge_index[0][e]
    unsigned p = atomicAdd(ecnt, 1u);
    pairs[2 * p]     = src;
    pairs[2 * p + 1] = dst - GEN_BASE;
    atomicAdd(&cnt[dst - GEN_BASE], 1u);
  }
}

// fully-unrolled 16-wide embedding for compile-time input width F
template <int F>
__device__ __forceinline__ void embed_fixed(const float* __restrict__ W,
                                            const float* __restrict__ b,
                                            const float* __restrict__ f,
                                            float* __restrict__ xj) {
  float fv[F];
#pragma unroll
  for (int k = 0; k < F; ++k) fv[k] = f[k];
#pragma unroll
  for (int o = 0; o < 16; ++o) {
    float acc = b[o];
#pragma unroll
    for (int k = 0; k < F; ++k) acc = fmaf(W[o * F + k], fv[k], acc);
    xj[o] = acc;
  }
}

// ---------------- K3: per-16-edge tile, conv via v_wmma_f32_16x16x32_bf16 ----------------
// Lanes 0-15 build branch-1 A rows (edge = lane), lanes 16-31 build branch-2 rows.
// Padded (invalid) slots carry zero A-rows and scatter to the dump row -> branch-free epilogue.
__global__ __launch_bounds__(256)
void edge_msg_wmma(const int* __restrict__ pairs, const unsigned* __restrict__ ecnt,
                   const float* __restrict__ xg1, const float* __restrict__ xg2,
                   float* __restrict__ sum1, float* __restrict__ sum2,
                   const float* __restrict__ p1, const float* __restrict__ p2,
                   const float* __restrict__ xbus, const float* __restrict__ xline,
                   const float* __restrict__ xload) {
  __shared__ __bf16    abuf[8][2][16][32];  // [wave][branch][edge-slot][k]
  __shared__ unsigned  dstl[8][16];         // precomputed dword offset: row*16 (or dump)

  const int lane = threadIdx.x & 31;
  const int w    = threadIdx.x >> 5;
  const int n16  = lane & 15;
  const int hi   = lane >> 4;               // 0: lanes 0-15 (branch 1), 1: lanes 16-31 (branch 2)

  // B fragments (Wc^T, 32x16 bf16): col = lane%16, K-half by lane/16, 2 K per VGPR
  v16bf B1, B2;
  {
    const int kb = hi ? 16 : 0;
#pragma unroll
    for (int i = 0; i < 16; ++i) {
      B1[i] = (__bf16)p1[P_CONV_W + n16 * 32 + kb + i];
      B2[i] = (__bf16)p2[P_CONV_W + n16 * 32 + kb + i];
    }
  }

  const unsigned count   = *ecnt;
  const unsigned ntiles  = (count + 15u) >> 4;
  const unsigned wid     = blockIdx.x * 8u + (unsigned)w;
  const unsigned wstride = gridDim.x * 8u;

  for (unsigned t = wid; t < ntiles; t += wstride) {
    const unsigned e = (t << 4) + (unsigned)n16;
    const bool valid = e < count;
    int src = 0, dl = 0;
    if (valid) { src = pairs[2 * e]; dl = pairs[2 * e + 1]; }
    if (hi == 0) dstl[w][n16] = valid ? (unsigned)dl * 16u : (unsigned)NGv * 16u;

    const float* pp = hi ? p2 : p1;
    const float* xg = hi ? xg2 : xg1;

    // xj = embed(src) on the fly (only ~1.5% of edges survive the filter);
    // each node-type branch is straight-line FMA code (compile-time F)
    float xj[16];
    if (src >= GEN_BASE) {
      const float* r = xg + (size_t)(src - GEN_BASE) * 16;
#pragma unroll
      for (int o = 0; o < 16; ++o) xj[o] = r[o];
    } else if (src < NBv) {
      embed_fixed<6>(pp + P_BUS_W,  pp + P_BUS_B,  xbus  + (size_t)src * 6, xj);
    } else if (src < NBv + NLv) {
      embed_fixed<8>(pp + P_LINE_W, pp + P_LINE_B, xline + (size_t)(src - NBv) * 8, xj);
    } else {
      embed_fixed<4>(pp + P_LOAD_W, pp + P_LOAD_B, xload + (size_t)(src - NBv - NLv) * 4, xj);
    }

    // stage A row: [xi, xj - xi] as bf16 (invalid edge -> zero row => exact 0.0 messages)
    const float* xi = xg + (size_t)dl * 16;
#pragma unroll
    for (int c = 0; c < 16; ++c) {
      float xic = valid ? xi[c] : 0.f;
      float dlt = valid ? (xj[c] - xic) : 0.f;
      abuf[w][hi][n16][c]      = (__bf16)xic;
      abuf[w][hi][n16][16 + c] = (__bf16)dlt;
    }

    // intra-wave cross-lane LDS visibility (CDNA5 split counters)
    asm volatile("s_wait_dscnt 0" ::: "memory");

    // assemble A fragments: contiguous K order (k = 16*(lane/16) + i), matching B
    v16bf A1, A2;
#pragma unroll
    for (int i = 0; i < 16; ++i) {
      const int k = i + (hi ? 16 : 0);
      A1[i] = abuf[w][0][n16][k];
      A2[i] = abuf[w][1][n16][k];
    }

    v8f c0 = {};
    v8f d1 = __builtin_amdgcn_wmma_f32_16x16x32_bf16(false, A1, false, B1, (short)0, c0, false, false);
    v8f d2 = __builtin_amdgcn_wmma_f32_16x16x32_bf16(false, A2, false, B2, (short)0, c0, false, false);

    // branch-free scatter: VGPR r holds row m = r + 8*hi (edge slot), col = lane%16 (channel).
    // The 8 offsets this lane needs are contiguous in LDS -> b128 loads.
    unsigned offs[8];
#pragma unroll
    for (int r = 0; r < 8; ++r) offs[r] = dstl[w][hi * 8 + r];
#pragma unroll
    for (int r = 0; r < 8; ++r) {
      atomicAdd(&sum1[(size_t)(offs[r] + n16)], d1[r]);
      atomicAdd(&sum2[(size_t)(offs[r] + n16)], d2[r]);
    }
  }
}

// ---------------- K4: h = relu((sum + cnt*b)/max(cnt,1)); value head 192->1 ----------------
__global__ void finalize_val(const float* __restrict__ sum1, const float* __restrict__ sum2,
                             const unsigned* __restrict__ cnt,
                             const float* __restrict__ xg1, const float* __restrict__ xg2,
                             const float* __restrict__ p1, const float* __restrict__ p2,
                             float* __restrict__ out) {
  int tid = blockIdx.x * blockDim.x + threadIdx.x;
  if (tid >= 2 * Gv) return;
  const int br = tid >> 12;
  const int g  = tid & (Gv - 1);
  const float* sum = br ? sum2 : sum1;
  const float* xg  = br ? xg2  : xg1;
  const float* pp  = br ? p2   : p1;
  const float* cb  = pp + P_CONV_B;
  const float* Wv  = pp + P_VAL_W;

  float acc = pp[P_VAL_B];
#pragma unroll
  for (int j = 0; j < 6; ++j) {
    const int n = g * 6 + j;
    const float cf  = (float)cnt[n];
    const float inv = 1.0f / fmaxf(cf, 1.0f);
#pragma unroll
    for (int c = 0; c < 16; ++c) {
      float h = (sum[n * 16 + c] + cf * cb[c]) * inv;   // bias folded: sum_nobias + cnt*b
      h = fmaxf(h, 0.0f);
      acc = fmaf(Wv[j * 32 + c], h, acc);
      acc = fmaf(Wv[j * 32 + 16 + c], xg[n * 16 + c], acc);  // skip
    }
  }
  out[br * Gv + g] = acc;
}

// ---------------- launcher ----------------
extern "C" void kernel_launch(void* const* d_in, const int* in_sizes, int n_in,
                              void* d_out, int out_size, void* d_ws, size_t ws_size,
                              hipStream_t stream) {
  const float* xbus   = (const float*)d_in[0];
  const float* xline  = (const float*)d_in[1];
  const float* xload  = (const float*)d_in[2];
  const float* xgen   = (const float*)d_in[3];
  const float* action = (const float*)d_in[4];
  const int*   ei     = (const int*)d_in[5];
  const float* p1     = (const float*)d_in[6];
  const float* p2     = (const float*)d_in[7];
  float* out = (float*)d_out;

  // workspace layout (floats): xg1 | xg2 | sum1 | sum2 | cnt | ecnt | pairs[2E]
  float* ws   = (float*)d_ws;
  float* xg1  = ws;
  float* xg2  = ws + (size_t)NGv * 16;
  float* sum1 = ws + (size_t)2 * NGv * 16;
  float* sum2 = sum1 + (size_t)SUMROWS * 16;
  unsigned* cnt  = (unsigned*)(sum2 + (size_t)SUMROWS * 16);
  unsigned* ecnt = cnt + NGv;
  int* pairs     = (int*)(ecnt + 1);

  // K0: zero sums (incl. dump rows) + counts + edge counter (contiguous from sum1)
  const int zn = 2 * SUMROWS * 16 + NGv + 1;
  ws_zero<<<(zn + 255) / 256, 256, 0, stream>>>((unsigned*)sum1, zn);

  // K1: gen embeddings (both branches)
  embed_gen<<<(NGv * 16 + 255) / 256, 256, 0, stream>>>(xgen, action, p1, p2, xg1, xg2);

  // K2: compact the ~1.5% of edges whose dst is a gen node
  edge_filter<<<(Ev + 255) / 256, 256, 0, stream>>>(ei, pairs, ecnt, cnt);

  // K3: WMMA message pass + branch-free atomic scatter (grid-strided waves over 16-edge tiles)
  edge_msg_wmma<<<512, 256, 0, stream>>>(pairs, ecnt, xg1, xg2, sum1, sum2,
                                         p1, p2, xbus, xline, xload);

  // K4: finalize mean/relu + value head
  finalize_val<<<(2 * Gv + 255) / 256, 256, 0, stream>>>(sum1, sum2, cnt, xg1, xg2, p1, p2, out);
}
